// minGRU_51539608287
// MI455X (gfx1250) — compile-verified, matrix operand order checked by reading, not run on hardware
//
#include <hip/hip_runtime.h>
#include <hip/hip_bf16.h>

// ---------------------------------------------------------------------------
// Problem constants (minGRU, B=4, S=4096, D=1024)
// ---------------------------------------------------------------------------
constexpr int Bb = 4;
constexpr int Ss = 4096;
constexpr int Dd = 1024;
constexpr int CH = 16;                     // scan chunks along S
constexpr int CL = Ss / CH;                // 256 steps per chunk
constexpr long NTOT = (long)Bb * Ss * Dd;  // 16,777,216

// GEMM tiling
constexpr int BM  = 128;                   // rows per workgroup
constexpr int BN  = 64;                    // hidden cols (and gate cols) per WG
constexpr int BK2 = 64;                    // K per TDM stage (2 WMMA K-steps)
constexpr unsigned A_BYTES = BM * BK2 * 2;           // 16 KB
constexpr unsigned B_BYTES = 2 * BN * BK2 * 2;       // 16 KB
constexpr unsigned STAGE   = A_BYTES + B_BYTES;      // 32 KB per buffer

typedef __attribute__((ext_vector_type(16))) __bf16 v16bf;
typedef __attribute__((ext_vector_type(8)))  float  v8f;
typedef __attribute__((ext_vector_type(4))) unsigned int u32x4;
typedef __attribute__((ext_vector_type(8))) int i32x8;
typedef __attribute__((ext_vector_type(4))) int i32x4;

union Frag { unsigned int u[8]; v16bf v; };

#if __has_builtin(__builtin_amdgcn_tensor_load_to_lds)
#define HAVE_TDM 1
#else
#define HAVE_TDM 0
#endif

// ---------------------------------------------------------------------------
// Helpers
// ---------------------------------------------------------------------------
__device__ __forceinline__ unsigned int bf16r(unsigned int u) {
  return (u + 0x7FFFu + ((u >> 16) & 1u)) >> 16;  // RNE fp32->bf16
}
__device__ __forceinline__ unsigned int pack_bf16(float a, float b) {
  return bf16r(__float_as_uint(a)) | (bf16r(__float_as_uint(b)) << 16);
}
__device__ __forceinline__ unsigned short bf16s(float a) {
  return (unsigned short)bf16r(__float_as_uint(a));
}

__device__ __forceinline__ float softplus_f(float x) {
  float ax = fabsf(x);
  return fmaxf(x, 0.0f) + __logf(1.0f + __expf(-ax));
}
__device__ __forceinline__ float log_g_f(float x) {
  return (x >= 0.0f) ? __logf(x + 0.5f) : -softplus_f(-x);
}
__device__ __forceinline__ float lse2(float m, float x) {
  float mx = fmaxf(m, x);
  float mn = fminf(m, x);
  float d  = mn - mx;
  if (!(d > -80.0f)) return mx;   // robust for -inf / NaN(-inf - -inf)
  return mx + __logf(1.0f + __expf(d));
}

__device__ __forceinline__ v16bf load_frag(const unsigned int* base) {
  Frag f;
  const uint4* p = (const uint4*)base;
  uint4 lo = p[0];
  uint4 hi = p[2];   // +8 uints = +16 bf16 (second K half of the fragment)
  f.u[0] = lo.x; f.u[1] = lo.y; f.u[2] = lo.z; f.u[3] = lo.w;
  f.u[4] = hi.x; f.u[5] = hi.y; f.u[6] = hi.z; f.u[7] = hi.w;
  return f.v;
}

__device__ __forceinline__ void wait_tensor0() {
#if __has_builtin(__builtin_amdgcn_s_wait_tensorcnt)
  __builtin_amdgcn_s_wait_tensorcnt(0);
#else
  asm volatile("s_wait_tensorcnt 0x0" ::: "memory");
#endif
}

// The TDM builtin carries the LDS destination only as integer bits inside the
// descriptor, so alias analysis thinks the shared buffer is never written and
// folds all fragment loads to undef. Escaping the pointer through an asm
// operand with a memory clobber forces the compiler to assume it was written.
__device__ __forceinline__ void lds_publish(const void* p) {
  asm volatile("" ::"v"(p) : "memory");
}

#if HAVE_TDM
// Loop-invariant part of the 2D tile descriptor (group 1).
__device__ __forceinline__ i32x8 tdm_g1(unsigned tile_d0, unsigned tile_d1,
                                        unsigned long stride0) {
  const unsigned td0 = 0x40000000u;                 // huge dims: no OOB clip
  const unsigned td1 = 0x00100000u;
  i32x8 g1;
  g1[0] = (int)(1u << 16);                          // data_size=1 (2B), mask=0
  g1[1] = (int)((td0 & 0xFFFFu) << 16);             // tensor_dim0[15:0]
  g1[2] = (int)(((td0 >> 16) & 0xFFFFu) | ((td1 & 0xFFFFu) << 16));
  g1[3] = (int)(((td1 >> 16) & 0xFFFFu) | (tile_d0 << 16));  // tile_dim0
  g1[4] = (int)(tile_d1 & 0xFFFFu);                 // tile_dim1 (tile_dim2=0)
  g1[5] = (int)(stride0 & 0xFFFFFFFFu);             // tensor_dim0_stride lo
  g1[6] = (int)((stride0 >> 32) & 0xFFFFu);         // stride hi (stride1=0)
  g1[7] = 0;
  return g1;
}

// Per-stage variable part (group 0) + issue.
__device__ __forceinline__ void tdm_issue(unsigned lds_off, unsigned long ga,
                                          i32x8 g1) {
  u32x4 g0;
  g0.x = 1u;                                        // count=1 (valid user D#)
  g0.y = lds_off;                                   // lds_addr (bytes)
  g0.z = (unsigned)(ga & 0xFFFFFFFFu);              // global_addr[31:0]
  g0.w = (unsigned)((ga >> 32) & 0x01FFFFFFu) | (2u << 30);  // addr[56:32]|type
  i32x4 z4 = {0, 0, 0, 0};
#if __has_include(<hip/amd_detail/amd_gfx1250_TDM.h>)
  i32x8 z8 = {0, 0, 0, 0, 0, 0, 0, 0};
  __builtin_amdgcn_tensor_load_to_lds(g0, g1, z4, z4, z8, 0);
#else
  __builtin_amdgcn_tensor_load_to_lds(g0, g1, z4, z4, 0);
#endif
}
#endif

// Fallback staging (no TDM builtin): plain vectorized copies of bf16 tiles.
__device__ __forceinline__ void stage_copy(unsigned char* lds,
                                           const unsigned short* xb,
                                           const unsigned short* wt,
                                           long rowbase, int colbase, int k0,
                                           int tid) {
  uint4* Ad = (uint4*)lds;
#pragma unroll
  for (int it = 0; it < 4; ++it) {
    int i = tid + it * 256;            // 1024 uint4 = 16 KB
    int r = i >> 3, q = i & 7;
    Ad[i] = *(const uint4*)(xb + (rowbase + r) * Dd + k0 + q * 8);
  }
  uint4* Bd = (uint4*)(lds + A_BYTES);
#pragma unroll
  for (int it = 0; it < 4; ++it) {
    int i = tid + it * 256;
    int r = i >> 3, q = i & 7;         // r = mat*64 + col
    long col = (r < BN) ? (long)(colbase + r) : (long)(Dd + colbase + (r - BN));
    Bd[i] = *(const uint4*)(wt + col * Dd + k0 + q * 8);
  }
}

// ---------------------------------------------------------------------------
// Kernel 0a: x (fp32) -> x_bf16 packed, 8 elems/thread.
// ---------------------------------------------------------------------------
__global__ __launch_bounds__(256) void convert_x(const float* __restrict__ x,
                                                 unsigned int* __restrict__ xb) {
  long t = (long)blockIdx.x * blockDim.x + threadIdx.x;
  const float4* p = (const float4*)(x + t * 8);
  float4 a = p[0], b = p[1];
  uint4 o;
  o.x = pack_bf16(a.x, a.y);
  o.y = pack_bf16(a.z, a.w);
  o.z = pack_bf16(b.x, b.y);
  o.w = pack_bf16(b.z, b.w);
  *(uint4*)(xb + t * 4) = o;
}

// ---------------------------------------------------------------------------
// Kernel 0b: W (D x 2D fp32, row-major) -> W^T bf16 (2D x D), LDS-tiled.
// ---------------------------------------------------------------------------
__global__ __launch_bounds__(256) void transpose_w(const float* __restrict__ W,
                                                   unsigned short* __restrict__ wt) {
  __shared__ unsigned short tile[32][33];
  int ct = blockIdx.x;                  // 0..63 over 2048 output rows (cols of W)
  int kt = blockIdx.y;                  // 0..31 over 1024 K
  int t = threadIdx.x;
#pragma unroll
  for (int i = 0; i < 4; ++i) {
    int k = (t >> 5) + i * 8;
    int c = t & 31;
    tile[k][c] = bf16s(W[(long)(kt * 32 + k) * (2 * Dd) + ct * 32 + c]);
  }
  __syncthreads();
  unsigned int* wo = (unsigned int*)wt;
#pragma unroll
  for (int i = 0; i < 2; ++i) {
    int c  = (t >> 4) + i * 16;
    int kk = (t & 15) * 2;
    unsigned int v = (unsigned)tile[kk][c] | ((unsigned)tile[kk + 1][c] << 16);
    wo[(((long)(ct * 32 + c)) * Dd + kt * 32 + kk) >> 1] = v;
  }
}

// ---------------------------------------------------------------------------
// Kernel 1: bf16 WMMA GEMM with TDM-staged, double-buffered LDS tiles +
// fused log-space epilogue. Wave w owns rows [w*16, w*16+16): 4 hidden +
// 4 gate 16x16 tiles over the same (M,N), so pairing is lane-aligned.
// waves_per_eu(1,4) relaxes the occupancy-driven VGPR budget so the
// scheduler can keep all 8 B fragments live and batch the LDS loads
// ahead of the WMMA burst (partial dscnt waits instead of full drains);
// 64 KB LDS/block still allows 5 blocks per WGP (320 KB).
// ---------------------------------------------------------------------------
__global__ __launch_bounds__(256)
__attribute__((amdgpu_waves_per_eu(1, 4)))
void gemm_fused(
    const unsigned short* __restrict__ xb, const unsigned short* __restrict__ wt,
    float* __restrict__ lc, float* __restrict__ lv) {
  __shared__ __align__(16) unsigned char lds_raw[2 * STAGE];  // 64 KB

  const int tid  = threadIdx.x;
  const int lane = tid & 31;
  const int wave = tid >> 5;
  const long rowbase = (long)blockIdx.x * BM;
  const int  colbase = blockIdx.y * BN;
  const int half = lane >> 4;
  const int kb2  = half * 4;            // uint offset of this lane's K base
  const int ln15 = lane & 15;

  v8f acc[8] = {};                      // [0..3] hidden, [4..7] gate
  const int nst = Dd / BK2;             // 16 stages

#if HAVE_TDM
  // Loop-invariant descriptor state (wave-uniform -> SGPRs).
  const i32x8 g1A = tdm_g1(BK2, BM, Dd);
  const i32x8 g1B = tdm_g1(BK2, BN, Dd);
  const unsigned long gaA0 = (unsigned long)(xb + rowbase * Dd);
  const unsigned long gaBh = (unsigned long)(wt + (long)colbase * Dd);
  const unsigned long gaBg = (unsigned long)(wt + (long)(colbase + Dd) * Dd);

  if (wave == 0) {
    tdm_issue(0, gaA0, g1A);
    tdm_issue(A_BYTES, gaBh, g1B);
    tdm_issue(A_BYTES + B_BYTES / 2, gaBg, g1B);
  }
  for (int st = 0; st < nst; ++st) {
    if (wave == 0) wait_tensor0();      // stage st landed in LDS
    __syncthreads();                    // visible to all; prev compute done
    lds_publish(lds_raw);               // compiler: LDS was written externally
    if (wave == 0 && st + 1 < nst) {    // overlap DMA(st+1) with compute(st)
      const unsigned long ko = (unsigned long)((st + 1) * BK2) * 2ul;
      const unsigned lb = ((unsigned)(st + 1) & 1u) * STAGE;
      tdm_issue(lb, gaA0 + ko, g1A);
      tdm_issue(lb + A_BYTES, gaBh + ko, g1B);
      tdm_issue(lb + A_BYTES + B_BYTES / 2, gaBg + ko, g1B);
    }
    const unsigned int* Ab  = (const unsigned int*)(lds_raw + (st & 1) * STAGE);
    const unsigned int* Bb2 = Ab + A_BYTES / 4;
#else
  for (int st = 0; st < nst; ++st) {
    stage_copy(lds_raw, xb, wt, rowbase, colbase, st * BK2, tid);
    __syncthreads();
    const unsigned int* Ab  = (const unsigned int*)lds_raw;
    const unsigned int* Bb2 = Ab + A_BYTES / 4;
#endif
#pragma unroll
    for (int ks = 0; ks < 2; ++ks) {
      const int ko = ks * 16 + kb2;
      v16bf fa = load_frag(Ab + (wave * 16 + ln15) * 32 + ko);
      v16bf fb[8];
#pragma unroll
      for (int ct = 0; ct < 4; ++ct) {
        fb[ct]     = load_frag(Bb2 + (ct * 16 + ln15) * 32 + ko);
        fb[4 + ct] = load_frag(Bb2 + (BN + ct * 16 + ln15) * 32 + ko);
      }
#pragma unroll
      for (int q = 0; q < 8; ++q)
        acc[q] = __builtin_amdgcn_wmma_f32_16x16x32_bf16(
            false, fa, false, fb[q], (short)0, acc[q], false, false);
    }
#if !HAVE_TDM
    __syncthreads();
#endif
  }

  // Fused epilogue: C layout lane l, VGPR i -> M = i + 8*(l>=16), N = l&15.
  const int moff = half * 8;
#pragma unroll
  for (int ct = 0; ct < 4; ++ct) {
#pragma unroll
    for (int i = 0; i < 8; ++i) {
      float hv = acc[ct][i];
      float gv = acc[4 + ct][i];
      float lcv = -softplus_f(gv);
      float lvv = -softplus_f(-gv) + log_g_f(hv);
      long row = rowbase + wave * 16 + moff + i;
      long o   = row * Dd + (colbase + ct * 16 + ln15);
      lc[o] = lcv;
      lv[o] = lvv;
    }
  }
}

// ---------------------------------------------------------------------------
// Kernel 2 (phase A): in-place local chunk scans, one thread per (b,chunk,d):
// 65,536 independent 256-step chains. lc <- local cumsum, lv <- local cumLSE.
// ---------------------------------------------------------------------------
__global__ __launch_bounds__(256) void scan_chunks(float* __restrict__ lc,
                                                   float* __restrict__ lv) {
  int tid = blockIdx.x * blockDim.x + threadIdx.x;
  int d  = tid & (Dd - 1);
  int bc = tid / Dd;
  int c  = bc & (CH - 1);
  int b  = bc / CH;
  long base = ((long)b * Ss + (long)c * CL) * Dd + d;
  float a = 0.0f;
  float m = -INFINITY;
  for (int t = 0; t < CL; ++t) {
    long idx = base + (long)t * Dd;
    if (t + 8 < CL) {                       // hide HBM latency on the chain
      __builtin_prefetch(lc + idx + 8 * Dd, 0, 0);
      __builtin_prefetch(lv + idx + 8 * Dd, 0, 0);
    }
    float lcv = lc[idx];
    float lvv = lv[idx];
    a += lcv;
    m = lse2(m, lvv - a);
    lc[idx] = a;
    lv[idx] = m;
  }
}

// ---------------------------------------------------------------------------
// Kernel 3 (phase B): scan the 16 chunk aggregates per (b,d).
//   G(0) = -inf ; G(c+1) = A_c + LSE(G(c), S_l_end(c))
// ---------------------------------------------------------------------------
__global__ __launch_bounds__(256) void chunk_prefix(
    const float* __restrict__ lc, const float* __restrict__ lv,
    float* __restrict__ G) {
  int tid = blockIdx.x * blockDim.x + threadIdx.x;
  int d = tid & (Dd - 1);
  int b = tid / Dd;
  float g = -INFINITY;
  for (int c = 0; c < CH; ++c) {
    G[((long)b * CH + c) * Dd + d] = g;
    long last = ((long)b * Ss + (long)c * CL + (CL - 1)) * Dd + d;
    float Ac  = lc[last];
    float Sle = lv[last];
    g = Ac + lse2(g, Sle);
  }
}

// ---------------------------------------------------------------------------
// Kernel 4 (phase C): fully parallel fixup, float4-vectorized.
//   out = exp(a_l + LSE(G_c, S_l))
// ---------------------------------------------------------------------------
__global__ __launch_bounds__(256) void finalize(
    const float* __restrict__ lc, const float* __restrict__ lv,
    const float* __restrict__ G, float* __restrict__ out) {
  long i4  = (long)blockIdx.x * blockDim.x + threadIdx.x;
  long idx = i4 * 4;
  int  d4  = (int)(idx & (Dd - 1));
  long bs  = idx / Dd;
  int  s   = (int)(bs & (Ss - 1));
  int  b   = (int)(bs / Ss);
  int  c   = s / CL;

  float4 al = *(const float4*)(lc + idx);
  float4 sl = *(const float4*)(lv + idx);
  float4 g4 = *(const float4*)(G + ((long)b * CH + c) * Dd + d4);

  float4 o;
  o.x = __expf(al.x + lse2(g4.x, sl.x));
  o.y = __expf(al.y + lse2(g4.y, sl.y));
  o.z = __expf(al.z + lse2(g4.z, sl.z));
  o.w = __expf(al.w + lse2(g4.w, sl.w));
  *(float4*)(out + idx) = o;
}

// ---------------------------------------------------------------------------
// Launcher
// ---------------------------------------------------------------------------
extern "C" void kernel_launch(void* const* d_in, const int* in_sizes, int n_in,
                              void* d_out, int out_size, void* d_ws, size_t ws_size,
                              hipStream_t stream) {
  const float* x = (const float*)d_in[0];   // (B,S,D) fp32
  const float* W = (const float*)d_in[1];   // (D, 2D) fp32
  float* out = (float*)d_out;               // (B,S,D) fp32

  float* lc = (float*)d_ws;                 // B*S*D floats (64 MB)
  float* lv = lc + NTOT;                    // B*S*D floats (64 MB)
  float* G  = lv + NTOT;                    // B*CH*D floats (256 KB)
  unsigned short* xbf = (unsigned short*)(G + (long)Bb * CH * Dd);  // 32 MB bf16
  unsigned short* wtb = xbf + NTOT;         // W^T bf16, 2048 x 1024 (4 MB)

  // 0) one-shot precision conversion (memory-bound, ~2 us total)
  convert_x<<<(int)(NTOT / (256 * 8)), 256, 0, stream>>>(x, (unsigned int*)xbf);
  transpose_w<<<dim3(2 * Dd / 32, Dd / 32), 256, 0, stream>>>(W, wtb);
  // 1) fused GEMM + elementwise log-space transforms (TDM-staged tiles)
  gemm_fused<<<dim3((Bb * Ss) / BM, Dd / BN), 256, 0, stream>>>(xbf, wtb, lc, lv);
  // 2) local chunk scans (in-place)
  scan_chunks<<<(Bb * CH * Dd) / 256, 256, 0, stream>>>(lc, lv);
  // 3) chunk-aggregate prefix per (b, d)
  chunk_prefix<<<(Bb * Dd) / 256, 256, 0, stream>>>(lc, lv, G);
  // 4) parallel fixup -> output
  finalize<<<(int)(NTOT / (256 * 4)), 256, 0, stream>>>(lc, lv, G, out);
}
